// LayerNormLSTMCell_74706661147025
// MI455X (gfx1250) — matrix operand order repaired and checked by reading, not compiled, not run on hardware
//
#include <hip/hip_runtime.h>
#include <cstdint>
#include <cstddef>

// ---------------------------------------------------------------------------
// LayerNorm LSTM cell for MI455X (gfx1250, wave32, WMMA).
//
// gates = [x|hx] (8192x2048) @ [W_ih|W_hh]^T (4096x2048) + b_ih + b_hh
// then per-gate LN + sigmoid/tanh, cy = LN(f*cx + i*c), hy = o*tanh(cy).
//
// Roofline: ~137 GFLOP, ~130-230MB HBM -> memory floor ~6-10us @23.3TB/s.
// bf16 WMMA (16x16x32, fp32 accum) keeps compute near that floor; fp32 WMMA
// (16x16x4) is ~8x slower per instruction and deeply compute bound.
// The 16MB bf16 weight matrix is L2-resident (192MB L2), so per-M-tile
// weight re-reads hit L2, not HBM.
//
// LDS staging of the A panel uses global_load_async_to_lds_b128 (ASYNCcnt),
// and the A fragments are read back with explicit ds_load_b128 inline asm:
// both ends of the LDS traffic are hardware-ISA, sequenced by
// s_wait_asynccnt / s_wait_dscnt / workgroup barriers, so the compiler
// cannot constant-fold the staged data away.
// ---------------------------------------------------------------------------

typedef __attribute__((ext_vector_type(16))) __bf16        v16bf;
typedef __attribute__((ext_vector_type(8)))  float         v8f;
typedef __attribute__((ext_vector_type(4)))  unsigned int  v4u;

#define HDIM 1024
#define BROWS 8192
#define NTOT 4096
#define KTOT 2048
#define KC   1024                // K-chunk staged in LDS
#define LDS_ROW (KC * 2 + 16)   // 2064B: 16B pad per row -> conflict-free ds_load_b128

static __device__ __forceinline__ unsigned short f32_to_bf16(float f) {
  union { float f; unsigned int u; } c; c.f = f;
  unsigned int u = c.u;
  u += 0x7FFFu + ((u >> 16) & 1u);   // round-to-nearest-even
  return (unsigned short)(u >> 16);
}

// Pack two fp32 matrices [rows x 1024] side-by-side into bf16 [rows x 2048].
__global__ void pack2_bf16_kernel(const float* __restrict__ a,
                                  const float* __restrict__ b,
                                  unsigned short* __restrict__ out,
                                  int rows) {
  int idx = blockIdx.x * blockDim.x + threadIdx.x;
  int row = idx >> 11;            // / 2048
  int k   = idx & (KTOT - 1);
  if (row >= rows) return;
  float v = (k < HDIM) ? a[(size_t)row * HDIM + k]
                       : b[(size_t)row * HDIM + (k - HDIM)];
  out[idx] = f32_to_bf16(v);
}

// Sum across the 16 lanes of this lane's half-wave (masks 1,2,4,8 stay in-half).
static __device__ __forceinline__ float halfsum16(float v) {
  v += __shfl_xor(v, 1);
  v += __shfl_xor(v, 2);
  v += __shfl_xor(v, 4);
  v += __shfl_xor(v, 8);
  return v;
}

__global__ __launch_bounds__(1024, 1)
void lstm_wmma_ln_kernel(const unsigned short* __restrict__ XH,   // [8192][2048] bf16
                         const unsigned short* __restrict__ W,    // [4096][2048] bf16
                         const float* __restrict__ bias_ih,
                         const float* __restrict__ bias_hh,
                         const float* __restrict__ cx,
                         const float* __restrict__ g_i,  const float* __restrict__ b_i,
                         const float* __restrict__ g_f,  const float* __restrict__ b_f,
                         const float* __restrict__ g_c,  const float* __restrict__ b_c,
                         const float* __restrict__ g_cy, const float* __restrict__ b_cy,
                         const float* __restrict__ g_o,  const float* __restrict__ b_o,
                         float* __restrict__ hy,
                         float* __restrict__ cy_out) {
  // 16-row A panel for one K-chunk, bf16, padded rows.
  __shared__ __align__(16) unsigned char As[16 * LDS_ROW];   // 33,024 B
  __shared__ float red[32][16][2];                           //  4,096 B

  const int tid   = threadIdx.x;
  const int wave  = tid >> 5;       // 0..31 : owns h in [32w, 32w+32) of every gate
  const int lane  = tid & 31;
  const int half  = lane >> 4;      // WMMA lane-half
  const int l16   = lane & 15;
  const int hbase = wave * 32;
  const int rowBase = blockIdx.x * 16;
  const unsigned long long xh_base = (unsigned long long)XH;
  // Raw LDS byte offset of As[0]: low 32 bits of the LDS-aperture flat
  // address are the DS offset (ISA 10.2 aperture mapping). Used by both the
  // async writes and the asm ds_loads, so the coordinate system is consistent
  // with the compiler-placed `red` array.
  const unsigned int asBase = (unsigned int)(uintptr_t)(void*)As;

  // 8 N-tiles: t -> gate (t>>1) in {i,f,c,o}, sub-tile (t&1).
  v8f acc[8];
#pragma unroll
  for (int t = 0; t < 8; ++t) acc[t] = (v8f){0.f,0.f,0.f,0.f,0.f,0.f,0.f,0.f};

  for (int kc = 0; kc < 2; ++kc) {
    // ---- async-stage A panel chunk: 16 rows x 2048B, 2048 x 16B transfers ----
#pragma unroll
    for (int i = 0; i < 2; ++i) {
      int idx = tid + i * 1024;
      int r   = idx >> 7;           // 0..15
      int c16 = idx & 127;          // 16B unit within row
      unsigned int ldsOff = asBase + (unsigned int)(r * LDS_ROW + c16 * 16);
      unsigned int goff   = (unsigned int)((((rowBase + r) * KTOT) + kc * KC) * 2 + c16 * 16);
      asm volatile("global_load_async_to_lds_b128 %0, %1, %2"
                   : : "v"(ldsOff), "v"(goff), "s"(xh_base) : "memory");
    }
    asm volatile("s_wait_asynccnt 0" : : : "memory");
    __syncthreads();

    if (kc == 0)  // warm L2/GL1 for the second A chunk
      __builtin_prefetch(&XH[(size_t)(rowBase + (tid >> 6)) * KTOT + KC + (size_t)(tid & 63) * 16], 0, 0);

    // ---- K loop: 32 per WMMA ----
#pragma unroll 2
    for (int k0 = 0; k0 < KC; k0 += 32) {
      union Frag { v4u q[2]; v16bf v; };
      // A frag (ISA 16-bit A layout): half 0 -> K {k0..k0+7, k0+16..k0+23},
      //                               half 1 -> K {k0+8..k0+15, k0+24..k0+31}
      // Explicit ds_load_b128 asm so the staged (asm-written) LDS data is a
      // real per-iteration dependency the compiler cannot hoist or fold.
      Frag afr;
      v4u a0, a1;
      unsigned int aaddr = asBase
                         + (unsigned int)(l16 * LDS_ROW + (k0 + 8 * half) * 2);
      asm volatile("ds_load_b128 %0, %2\n\t"
                   "ds_load_b128 %1, %2 offset:32\n\t"
                   "s_wait_dscnt 0x0"
                   : "=v"(a0), "=v"(a1) : "v"(aaddr) : "memory");
      afr.q[0] = a0;
      afr.q[1] = a1;
#pragma unroll
      for (int t = 0; t < 8; ++t) {
        int g    = t >> 1;
        int ncol = g * HDIM + hbase + (t & 1) * 16 + l16;   // N = lane&15
        const unsigned short* wp = W + (size_t)ncol * KTOT + kc * KC + k0 + 8 * half;
        Frag bfr;                                           // B: K = {8*half..+7, 16+8*half..+7}, table order
        bfr.q[0] = *(const v4u*)(wp);
        bfr.q[1] = *(const v4u*)(wp + 16);
        acc[t] = __builtin_amdgcn_wmma_f32_16x16x32_bf16(
            false, afr.v, false, bfr.v, (short)0, acc[t], false, false);
      }
    }
    __syncthreads();   // all waves done reading As before restage
  }

  // ---- epilogue: acc[t][r] = gates(row = rowBase + r + 8*half, col = ncol(t)+l16) ----
  // fold the two bias vectors (column-dependent, so must precede LN)
#pragma unroll
  for (int t = 0; t < 8; ++t) {
    int col = (t >> 1) * HDIM + hbase + (t & 1) * 16 + l16;
    float bsum = bias_ih[col] + bias_hh[col];
#pragma unroll
    for (int r = 0; r < 8; ++r) acc[t][r] += bsum;
  }

  const float* gam[4] = {g_i, g_f, g_c, g_o};
  const float* bet[4] = {b_i, b_f, b_c, b_o};
  float mean_r[8], rstd_r[8];

  // per-gate LayerNorm over H=1024 (spread over all 32 waves) + activation
#pragma unroll
  for (int g = 0; g < 4; ++g) {
    __syncthreads();
#pragma unroll
    for (int r = 0; r < 8; ++r) {
      float s  = acc[2*g][r] + acc[2*g+1][r];
      float ss = acc[2*g][r] * acc[2*g][r] + acc[2*g+1][r] * acc[2*g+1][r];
      s = halfsum16(s); ss = halfsum16(ss);
      if (l16 == 0) { red[wave][r + 8*half][0] = s; red[wave][r + 8*half][1] = ss; }
    }
    __syncthreads();
#pragma unroll
    for (int r = 0; r < 8; ++r) {
      float s = 0.f, ss = 0.f;
      for (int wv = 0; wv < 32; ++wv) {
        s  += red[wv][r + 8*half][0];
        ss += red[wv][r + 8*half][1];
      }
      float m   = s * (1.0f / 1024.0f);
      float var = ss * (1.0f / 1024.0f) - m * m;
      mean_r[r] = m;
      rstd_r[r] = rsqrtf(var + 1e-5f);
    }
#pragma unroll
    for (int p = 0; p < 2; ++p) {
      int t = 2*g + p;
      int h = hbase + p * 16 + l16;
      float gg = gam[g][h], bb = bet[g][h];
#pragma unroll
      for (int r = 0; r < 8; ++r) {
        float xn = (acc[t][r] - mean_r[r]) * rstd_r[r] * gg + bb;
        acc[t][r] = (g == 2) ? tanhf(xn) : 1.0f / (1.0f + expf(-xn));
      }
    }
  }

  // u = f*cx + i*c   (all gates wave-local by construction)
  float u[2][8];
#pragma unroll
  for (int p = 0; p < 2; ++p) {
    int h = hbase + p * 16 + l16;
#pragma unroll
    for (int r = 0; r < 8; ++r) {
      int row = rowBase + r + 8 * half;
      float cxv = cx[(size_t)row * HDIM + h];
      u[p][r] = acc[2+p][r] * cxv + acc[0+p][r] * acc[4+p][r];
    }
  }

  // LayerNorm over cy, then outputs
  __syncthreads();
#pragma unroll
  for (int r = 0; r < 8; ++r) {
    float s  = u[0][r] + u[1][r];
    float ss = u[0][r]*u[0][r] + u[1][r]*u[1][r];
    s = halfsum16(s); ss = halfsum16(ss);
    if (l16 == 0) { red[wave][r + 8*half][0] = s; red[wave][r + 8*half][1] = ss; }
  }
  __syncthreads();
#pragma unroll
  for (int r = 0; r < 8; ++r) {
    float s = 0.f, ss = 0.f;
    for (int wv = 0; wv < 32; ++wv) {
      s  += red[wv][r + 8*half][0];
      ss += red[wv][r + 8*half][1];
    }
    float m   = s * (1.0f / 1024.0f);
    float var = ss * (1.0f / 1024.0f) - m * m;
    mean_r[r] = m;
    rstd_r[r] = rsqrtf(var + 1e-5f);
  }
#pragma unroll
  for (int p = 0; p < 2; ++p) {
    int h = hbase + p * 16 + l16;
    float gg = g_cy[h], bb = b_cy[h];
#pragma unroll
    for (int r = 0; r < 8; ++r) {
      int row = rowBase + r + 8 * half;
      float cyn = (u[p][r] - mean_r[r]) * rstd_r[r] * gg + bb;
      cy_out[(size_t)row * HDIM + h] = cyn;
      hy[(size_t)row * HDIM + h]     = acc[6+p][r] * tanhf(cyn);
    }
  }
}

extern "C" void kernel_launch(void* const* d_in, const int* in_sizes, int n_in,
                              void* d_out, int out_size, void* d_ws, size_t ws_size,
                              hipStream_t stream) {
  const float* x    = (const float*)d_in[0];
  const float* hx   = (const float*)d_in[1];
  const float* cx   = (const float*)d_in[2];
  const float* w_ih = (const float*)d_in[3];
  const float* w_hh = (const float*)d_in[4];
  const float* b_ih = (const float*)d_in[5];
  const float* b_hh = (const float*)d_in[6];
  const float* g_i  = (const float*)d_in[7];
  const float* bi_  = (const float*)d_in[8];
  const float* g_f  = (const float*)d_in[9];
  const float* bf_  = (const float*)d_in[10];
  const float* g_c  = (const float*)d_in[11];   // ln_cx_*
  const float* bc_  = (const float*)d_in[12];
  const float* g_cy = (const float*)d_in[13];   // ln_cy_*
  const float* bcy_ = (const float*)d_in[14];
  const float* g_o  = (const float*)d_in[15];
  const float* bo_  = (const float*)d_in[16];

  unsigned short* XHbf = (unsigned short*)d_ws;                                     // 32 MB
  unsigned short* Wbf  = (unsigned short*)((char*)d_ws + (size_t)BROWS * KTOT * 2); // 16 MB

  // fp32 -> bf16 packing (K-concatenation of the two GEMMs)
  pack2_bf16_kernel<<<(BROWS * KTOT) / 256, 256, 0, stream>>>(x, hx, XHbf, BROWS);
  pack2_bf16_kernel<<<(NTOT * KTOT) / 256, 256, 0, stream>>>(w_ih, w_hh, Wbf, NTOT);

  float* out = (float*)d_out;
  float* hy  = out;                            // outputs: (hy, cy) concatenated
  float* cy  = out + (size_t)BROWS * HDIM;

  lstm_wmma_ln_kernel<<<BROWS / 16, 1024, 0, stream>>>(
      XHbf, Wbf, b_ih, b_hh, cx,
      g_i, bi_, g_f, bf_, g_c, bc_, g_cy, bcy_, g_o, bo_,
      hy, cy);
}
